// Graph_25409026523559
// MI455X (gfx1250) — matrix operand order repaired and checked by reading, not compiled, not run on hardware
//
#include <hip/hip_runtime.h>
#include <hip/hip_bf16.h>

// Label propagation: W = exp(-pairdist/2) via WMMA Gram, top-k mask, sym-norm,
// Chebyshev solve of (I - alpha*S) F = L using bf16 hi/lo split WMMA matvecs.
// Hot matvec stages the S operand strip into LDS via the Tensor Data Mover.

#define NN      4096
#define DIM     256
#define NC      64
#define NSUP    1024
#define KTOP    20
#define CHEB_T  128
#define EPSV    2.220446049250313e-16f

// ---- k_spmm LDS staging geometry -----------------------------------------
#define KC      1024                  // K elements per TDM chunk
#define ROWB    2064                  // padded LDS row stride: 1024*2B + 2 pads * 8B
#define ABUF    (ROWB * 16)           // one bf16 plane: 33024 B
#define SMEMB   (2 * ABUF)            // hi + lo planes: 66048 B

typedef __bf16 v16bf __attribute__((ext_vector_type(16)));
typedef __bf16 bf8v  __attribute__((ext_vector_type(8)));
typedef float  v8f   __attribute__((ext_vector_type(8)));
typedef unsigned int u32x4 __attribute__((ext_vector_type(4)));
typedef int          i32x8 __attribute__((ext_vector_type(8)));
typedef int          i32x4 __attribute__((ext_vector_type(4)));

__device__ __forceinline__ float wave_sum_f(float v) {
#pragma unroll
  for (int off = 16; off > 0; off >>= 1) v += __shfl_xor(v, off, 32);
  return v;
}
__device__ __forceinline__ int wave_sum_i(int v) {
#pragma unroll
  for (int off = 16; off > 0; off >>= 1) v += __shfl_xor(v, off, 32);
  return v;
}

// Build a v16bf WMMA operand from two contiguous 8-element (16B) chunks.
__device__ __forceinline__ v16bf load16(const __bf16* p0, const __bf16* p1) {
  bf8v a = *(const bf8v*)p0;
  bf8v b = *(const bf8v*)p1;
  return __builtin_shufflevector(a, b, 0, 1, 2, 3, 4, 5, 6, 7,
                                 8, 9, 10, 11, 12, 13, 14, 15);
}

// ---------- Stage 1: e = emb/(sigma+eps)^2, split into bf16 hi + lo ----------
__global__ void k_scale_split(const float* __restrict__ emb,
                              const float* __restrict__ sigma,
                              __bf16* __restrict__ ehi, __bf16* __restrict__ elo) {
  int idx = blockIdx.x * blockDim.x + threadIdx.x;
  if (idx >= NN * DIM) return;
  int i = idx >> 8;  // / DIM
  float s = sigma[i] + EPSV;
  float v = emb[idx] / s;
  v = v / s;
  __bf16 h = (__bf16)v;
  ehi[idx] = h;
  elo[idx] = (__bf16)(v - (float)h);
}

// ---------- Stage 2: row squared norms (of the value WMMA will see) ----------
__global__ void k_sqnorm(const __bf16* __restrict__ ehi,
                         const __bf16* __restrict__ elo, float* __restrict__ sq) {
  int wave = threadIdx.x >> 5, lane = threadIdx.x & 31;
  int row = blockIdx.x * (blockDim.x >> 5) + wave;
  if (row >= NN) return;
  const __bf16* ph = ehi + (size_t)row * DIM;
  const __bf16* pl = elo + (size_t)row * DIM;
  float s = 0.f;
  for (int j = lane; j < DIM; j += 32) {
    float v = (float)ph[j] + (float)pl[j];
    s += v * v;
  }
  s = wave_sum_f(s);
  if (lane == 0) sq[row] = s;
}

// ---------- Stage 3: W = exp(-((|a|^2+|b|^2-2ab)/d)/2), WMMA Gram ----------
__global__ void k_gram_w(const __bf16* __restrict__ ehi,
                         const __bf16* __restrict__ elo,
                         const float* __restrict__ sq, float* __restrict__ W) {
  int lane = threadIdx.x & 31;
  int wv = threadIdx.x >> 5;
  int i0 = blockIdx.x * 16;
  int j0 = (blockIdx.y * 4 + wv) * 16;
  int h = lane >> 4, n = lane & 15;
  const __bf16* arh = ehi + (size_t)(i0 + n) * DIM;  // A row m (= n)
  const __bf16* arl = elo + (size_t)(i0 + n) * DIM;
  const __bf16* brh = ehi + (size_t)(j0 + n) * DIM;  // B column n = e row j0+n
  const __bf16* brl = elo + (size_t)(j0 + n) * DIM;
  v8f acc = {};
  for (int kb = 0; kb < DIM; kb += 32) {
    v16bf ah = load16(arh + kb + 8 * h, arh + kb + 16 + 8 * h);
    v16bf al = load16(arl + kb + 8 * h, arl + kb + 16 + 8 * h);
    v16bf bh = load16(brh + kb + 8 * h, brh + kb + 16 + 8 * h);
    v16bf bl = load16(brl + kb + 8 * h, brl + kb + 16 + 8 * h);
    acc = __builtin_amdgcn_wmma_f32_16x16x32_bf16(false, ah, false, bh, (short)0, acc, false, false);
    acc = __builtin_amdgcn_wmma_f32_16x16x32_bf16(false, ah, false, bl, (short)0, acc, false, false);
    acc = __builtin_amdgcn_wmma_f32_16x16x32_bf16(false, al, false, bh, (short)0, acc, false, false);
  }
  float sqj = sq[j0 + n];
#pragma unroll
  for (int r = 0; r < 8; ++r) {
    int i = i0 + 8 * h + r;
    float dist = (sq[i] + sqj - 2.0f * acc[r]) * (1.0f / DIM);
    W[(size_t)i * NN + (j0 + n)] = __expf(-0.5f * dist);
  }
}

// ---------- Stage 4: per-row top-20 threshold (bisection) + scatter mask ----
__global__ void k_topk_mask(const float* __restrict__ W,
                            unsigned char* __restrict__ mask) {
  int wave = threadIdx.x >> 5, lane = threadIdx.x & 31;
  int row = blockIdx.x * (blockDim.x >> 5) + wave;
  if (row >= NN) return;
  const float* wr = W + (size_t)row * NN;
  float lo = 0.f, hi = 2.f;  // invariant: cnt(>lo) > K, cnt(>hi) <= K
  for (int it = 0; it < 32; ++it) {
    float mid = 0.5f * (lo + hi);
    int cnt = 0;
    for (int j = lane; j < NN; j += 32) cnt += (wr[j] > mid) ? 1 : 0;
    cnt = wave_sum_i(cnt);
    if (cnt > KTOP) lo = mid; else hi = mid;
  }
  float th = hi;
  for (int j = lane; j < NN; j += 32) {
    if (wr[j] > th) {
      mask[(size_t)row * NN + j] = 1;  // duplicate writes of identical value:
      mask[(size_t)j * NN + row] = 1;  // benign + deterministic
    }
  }
}

// ---------- Stage 5: degrees (row sums of symmetric W*mask) -> D^-1/2 -------
__global__ void k_degree(const float* __restrict__ W,
                         const unsigned char* __restrict__ mask,
                         float* __restrict__ dinv) {
  int wave = threadIdx.x >> 5, lane = threadIdx.x & 31;
  int row = blockIdx.x * (blockDim.x >> 5) + wave;
  if (row >= NN) return;
  const float* wr = W + (size_t)row * NN;
  const unsigned char* mr = mask + (size_t)row * NN;
  float s = 0.f;
  for (int j = lane; j < NN; j += 32) s += mr[j] ? wr[j] : 0.f;
  s = wave_sum_f(s);
  if (lane == 0) dinv[row] = sqrtf(1.0f / (s + EPSV));
}

// ---------- Stage 6: S = dinv_i * (W o mask) * dinv_j, split hi/lo bf16 -----
__global__ void k_make_s(const float* __restrict__ W,
                         const unsigned char* __restrict__ mask,
                         const float* __restrict__ dinv,
                         __bf16* __restrict__ shi, __bf16* __restrict__ slo) {
  size_t idx = (size_t)blockIdx.x * blockDim.x + threadIdx.x;
  if (idx >= (size_t)NN * NN) return;
  int i = (int)(idx >> 12), j = (int)(idx & (NN - 1));
  float s = mask[idx] ? dinv[i] * W[idx] * dinv[j] : 0.f;
  __bf16 h = (__bf16)s;
  shi[idx] = h;
  slo[idx] = (__bf16)(s - (float)h);
}

// ---------- Stage 7: Chebyshev scalar coefficient sequence (on device) ------
__global__ void k_coeffs(const float* __restrict__ alpha,
                         float* __restrict__ c1, float* __restrict__ c2) {
  if (threadIdx.x != 0 || blockIdx.x != 0) return;
  float delta = alpha[0];              // spectrum of I-aS in [1-a, 1+a]; theta=1
  float sigma1 = 1.0f / delta;         // theta/delta
  float rho = delta;                   // rho_0 = 1/sigma1
  for (int k = 0; k < CHEB_T; ++k) {
    float rho_n = 1.0f / (2.0f * sigma1 - rho);
    c1[k] = rho_n * rho;
    c2[k] = 2.0f * rho_n / delta;
    rho = rho_n;
  }
}

// ---------- Stage 8: init x=0, r=b, d=b (theta=1); dT = split(d) ------------
__global__ void k_init(const int* __restrict__ labels, float* __restrict__ x,
                       float* __restrict__ r, float* __restrict__ dv,
                       __bf16* __restrict__ dThi, __bf16* __restrict__ dTlo) {
  int idx = blockIdx.x * blockDim.x + threadIdx.x;
  if (idx >= NN * NC) return;
  int i = idx >> 6, c = idx & 63;
  float b = (i < NSUP && labels[i] == c) ? 1.f : 0.f;
  x[idx] = 0.f;
  r[idx] = b;
  dv[idx] = b;
  __bf16 h = (__bf16)b;
  dThi[(size_t)c * NN + i] = h;
  dTlo[(size_t)c * NN + i] = (__bf16)(b - (float)h);
}

// ---- TDM: load a 16 x KC bf16 tile (row stride NN) into padded LDS ---------
// D# per CDNA5 ISA ch.8: group0 = {count, lds_addr, global_addr, type=2};
// group1 = {data_size=2B, pad 8B per 1024B, dims 4096, tile KC x 16, stride 4096}.
// This toolchain's builtin takes 6 args (g0 v4u, g1 v8i, g2 v4i, g3 v4i, v8i, cpol).
__device__ __forceinline__ void tdm_load_tile(unsigned int lds_addr, const void* gptr) {
#if __has_builtin(__builtin_amdgcn_tensor_load_to_lds)
  unsigned long long ga = (unsigned long long)gptr;
  u32x4 g0 = { 1u,                                   // count=1 (user mode)
               lds_addr,                             // LDS byte address
               (unsigned int)ga,                     // global_addr[31:0]
               (unsigned int)((ga >> 32) & 0x01FFFFFFu) | (2u << 30) };  // type=2
  i32x8 g1 = { (int)((1u << 16) | (1u << 20) | (7u << 22) | (1u << 25)),
               //   data_size=2B | pad_enable | interval=256DW | amount=2DW
               (int)0x10000000,        // tensor_dim0 = 4096 -> bits[79:48]
               (int)0x10000000,        // tensor_dim1 = 4096 -> bits[111:80]
               (int)(KC << 16),        // tile_dim0 = KC -> bits[127:112]
               16,                     // tile_dim1 = 16 rows
               NN,                     // tensor_dim0_stride = 4096
               0, 0 };
  i32x4 zz4 = { 0, 0, 0, 0 };          // groups 2/3 unused (2-D tensor)
  i32x8 zz8 = { 0, 0, 0, 0, 0, 0, 0, 0 };
  __builtin_amdgcn_tensor_load_to_lds(g0, g1, zz4, zz4, zz8, 0);
  __builtin_amdgcn_s_wait_tensorcnt(0);
#else
  // Fallback: cooperative wave copy with identical padded layout.
  extern __shared__ char smem_fb[];
  char* dst = smem_fb + (lds_addr - (unsigned int)(size_t)(void*)smem_fb);
  const __bf16* src = (const __bf16*)gptr;
  for (int t = threadIdx.x & 31; t < 16 * (KC / 8); t += 32) {
    int row = t / (KC / 8);
    int d = (t % (KC / 8)) * 16;           // data byte offset in row
    int p = d + ((d >> 10) << 3);          // padded offset
    *(bf8v*)(dst + row * ROWB + p) = *(const bf8v*)(src + (size_t)row * NN + d / 2);
  }
#endif
}

// ---------- Stage 9: y = S*d via WMMA; A strip staged in LDS by TDM ---------
// Block = 4 waves sharing one i-tile (16 rows of S); wave wv owns c-tile wv.
__global__ void k_spmm(const __bf16* __restrict__ shi, const __bf16* __restrict__ slo,
                       const __bf16* __restrict__ dThi, const __bf16* __restrict__ dTlo,
                       float* __restrict__ y) {
  extern __shared__ char smem[];
  char* sa_h = smem;
  char* sa_l = smem + ABUF;
  int lane = threadIdx.x & 31;
  int wv = threadIdx.x >> 5;
  int i0 = blockIdx.x * 16;
  int c0 = wv * 16;
  int h = lane >> 4, n = lane & 15;
  const __bf16* bh = dThi + (size_t)(c0 + n) * NN;
  const __bf16* bl = dTlo + (size_t)(c0 + n) * NN;
  const char* rowh = sa_h + n * ROWB;
  const char* rowl = sa_l + n * ROWB;
  unsigned int lds_h = (unsigned int)(size_t)(void*)sa_h;  // low 32 bits = LDS offset
  unsigned int lds_l = (unsigned int)(size_t)(void*)sa_l;
  v8f acc = {};
  for (int kc = 0; kc < NN; kc += KC) {
    __syncthreads();                 // previous chunk fully consumed
    if (threadIdx.x < 32) {          // wave 0 drives the TDM (TENSORcnt is per-wave)
      tdm_load_tile(lds_h, shi + (size_t)i0 * NN + kc);
      tdm_load_tile(lds_l, slo + (size_t)i0 * NN + kc);
    }
    __syncthreads();                 // publish LDS tile to all 4 waves
    for (int kb = 0; kb < KC; kb += 32) {
      int d0 = (kb + 8 * h) * 2;
      int d1 = (kb + 16 + 8 * h) * 2;
      int p0 = d0 + ((d0 >> 10) << 3);   // skip 8B pad per 1024B of data
      int p1 = d1 + ((d1 >> 10) << 3);
      v16bf a_h = load16((const __bf16*)(rowh + p0), (const __bf16*)(rowh + p1));
      v16bf a_l = load16((const __bf16*)(rowl + p0), (const __bf16*)(rowl + p1));
      int kg = kc + kb;
      v16bf b_h = load16(bh + kg + 8 * h, bh + kg + 16 + 8 * h);
      v16bf b_l = load16(bl + kg + 8 * h, bl + kg + 16 + 8 * h);
      acc = __builtin_amdgcn_wmma_f32_16x16x32_bf16(false, a_h, false, b_h, (short)0, acc, false, false);
      acc = __builtin_amdgcn_wmma_f32_16x16x32_bf16(false, a_h, false, b_l, (short)0, acc, false, false);
      acc = __builtin_amdgcn_wmma_f32_16x16x32_bf16(false, a_l, false, b_h, (short)0, acc, false, false);
    }
  }
#pragma unroll
  for (int r = 0; r < 8; ++r)
    y[(size_t)(i0 + 8 * h + r) * NC + (c0 + n)] = acc[r];
}

// ---------- Stage 10: Chebyshev update ------------------------------------
__global__ void k_cheb(const float* __restrict__ alpha, const float* __restrict__ c1a,
                       const float* __restrict__ c2a, int k,
                       const float* __restrict__ y, float* __restrict__ x,
                       float* __restrict__ r, float* __restrict__ dv,
                       __bf16* __restrict__ dThi, __bf16* __restrict__ dTlo) {
  int idx = blockIdx.x * blockDim.x + threadIdx.x;
  if (idx >= NN * NC) return;
  int i = idx >> 6, c = idx & 63;
  float al = alpha[0];
  float d_ = dv[idx];
  float Ad = d_ - al * y[idx];
  x[idx] += d_;
  float r_ = r[idx] - Ad;
  r[idx] = r_;
  float dn = c1a[k] * d_ + c2a[k] * r_;
  dv[idx] = dn;
  __bf16 h = (__bf16)dn;
  dThi[(size_t)c * NN + i] = h;
  dTlo[(size_t)c * NN + i] = (__bf16)(dn - (float)h);
}

extern "C" void kernel_launch(void* const* d_in, const int* in_sizes, int n_in,
                              void* d_out, int out_size, void* d_ws, size_t ws_size,
                              hipStream_t stream) {
  const float* emb    = (const float*)d_in[0];   // [4096,256]
  const float* sigma  = (const float*)d_in[1];   // [4096,1]
  const float* alpha  = (const float*)d_in[2];   // [1]
  const int*   labels = (const int*)d_in[3];     // [1024]
  // d_in[4] = alpha_k (=20), compile-time KTOP

  char* ws = (char*)d_ws;
  size_t off = 0;
  auto alloc = [&](size_t bytes) -> void* {
    off = (off + 255) & ~(size_t)255;
    void* p = ws + off;
    off += bytes;
    return p;
  };

  __bf16* ehi  = (__bf16*)alloc((size_t)NN * DIM * 2);
  __bf16* elo  = (__bf16*)alloc((size_t)NN * DIM * 2);
  float*  sq   = (float*) alloc((size_t)NN * 4);
  float*  W    = (float*) alloc((size_t)NN * NN * 4);
  unsigned char* mask = (unsigned char*)alloc((size_t)NN * NN);
  float*  dinv = (float*) alloc((size_t)NN * 4);
  __bf16* shi  = (__bf16*)alloc((size_t)NN * NN * 2);
  __bf16* slo  = (__bf16*)alloc((size_t)NN * NN * 2);
  float*  rv   = (float*) alloc((size_t)NN * NC * 4);
  float*  dvv  = (float*) alloc((size_t)NN * NC * 4);
  float*  yv   = (float*) alloc((size_t)NN * NC * 4);
  __bf16* dThi = (__bf16*)alloc((size_t)NC * NN * 2);
  __bf16* dTlo = (__bf16*)alloc((size_t)NC * NN * 2);
  float*  c1   = (float*) alloc(CHEB_T * 4);
  float*  c2   = (float*) alloc(CHEB_T * 4);
  float*  xout = (float*)d_out;                  // F accumulates here (4096x64)

  (void)hipMemsetAsync(mask, 0, (size_t)NN * NN, stream);

  k_scale_split<<<(NN * DIM) / 256, 256, 0, stream>>>(emb, sigma, ehi, elo);
  k_sqnorm<<<NN / 8, 256, 0, stream>>>(ehi, elo, sq);
  k_gram_w<<<dim3(NN / 16, NN / 64), 128, 0, stream>>>(ehi, elo, sq, W);
  k_topk_mask<<<NN / 8, 256, 0, stream>>>(W, mask);
  k_degree<<<NN / 8, 256, 0, stream>>>(W, mask, dinv);
  k_make_s<<<(int)(((size_t)NN * NN) / 256), 256, 0, stream>>>(W, mask, dinv, shi, slo);
  k_coeffs<<<1, 1, 0, stream>>>(alpha, c1, c2);
  k_init<<<(NN * NC) / 256, 256, 0, stream>>>(labels, xout, rv, dvv, dThi, dTlo);

  for (int k = 0; k < CHEB_T; ++k) {
    k_spmm<<<NN / 16, 128, SMEMB, stream>>>(shi, slo, dThi, dTlo, yv);
    k_cheb<<<(NN * NC) / 256, 256, 0, stream>>>(alpha, c1, c2, k, yv, xout, rv, dvv, dThi, dTlo);
  }
}
